// HybridQLSTM_65481071395969
// MI455X (gfx1250) — matrix operand order repaired and checked by reading, not compiled, not run on hardware
//
#include <hip/hip_runtime.h>
#include <hip/hip_bf16.h>
#include <math.h>

// ---------------- problem dims ----------------
#define T_STEPS 1024
#define BATCH   64
#define D_IN    256
#define HID     256
#define D_CAT   512          // D_IN + HID
#define NGATES  4
#define BH      (BATCH * HID)          // 16384

// WMMA tiling: 16x16x32 bf16
#define KT1 16               // D_CAT / 32
#define KT2 8                // HID  / 32
#define NT  16               // HID  / 16
#define MT  4                // BATCH/ 16

typedef __attribute__((ext_vector_type(16))) __bf16 v16bf;
typedef __attribute__((ext_vector_type(8)))  __bf16 v8bf;
typedef __attribute__((ext_vector_type(8)))  float  v8f;

// ---------------- helpers ----------------
__device__ __forceinline__ __bf16 f2bf(float f) {
    union { float f; unsigned u; } in; in.f = f;
    unsigned x = in.u;
    unsigned r = x + 0x7FFFu + ((x >> 16) & 1u);   // round-to-nearest-even
    union { unsigned short s; __bf16 b; } out;
    out.s = (unsigned short)(r >> 16);
    return out.b;
}

// A-fragment: 16 contiguous bf16 at p (8 at p, 8 at p+8 elems)
__device__ __forceinline__ v16bf ld_frag_a(const __bf16* p) {
    v8bf lo = *(const v8bf*)(p);
    v8bf hi = *(const v8bf*)(p + 8);
    v16bf r;
#pragma unroll
    for (int i = 0; i < 8; ++i) { r[i] = lo[i]; r[i + 8] = hi[i]; }
    return r;
}

__device__ __forceinline__ v8f wmma_bf16(v16bf a, v16bf b, v8f c) {
    return __builtin_amdgcn_wmma_f32_16x16x32_bf16(
        /*neg_a=*/false, a, /*neg_b=*/false, b,
        /*c_mod=*/(short)0, c, /*reuse_a=*/false, /*reuse_b=*/false);
}

__device__ __forceinline__ float fsigmoid(float x) {
    return 1.0f / (1.0f + __expf(-x));
}

// async-copy one 16B chunk global -> LDS (CDNA5, ASYNCcnt-tracked)
__device__ __forceinline__ void async_g2l_b128(unsigned lds_addr, unsigned goff,
                                               const void* gbase) {
    asm volatile("global_load_async_to_lds_b128 %0, %1, %2"
                 :: "v"(lds_addr), "v"(goff), "s"(gbase)
                 : "memory");
}
__device__ __forceinline__ void wait_async0() {
    asm volatile("s_wait_asynccnt 0x0" ::: "memory");
}

// device-scope grid barrier (counter monotonic within one launch; reset by prologue)
__device__ __forceinline__ void grid_sync(unsigned* bar, unsigned expected) {
    __threadfence();
    __syncthreads();
    if (threadIdx.x == 0) {
        __hip_atomic_fetch_add(bar, 1u, __ATOMIC_ACQ_REL, __HIP_MEMORY_SCOPE_AGENT);
        while (__hip_atomic_load(bar, __ATOMIC_ACQUIRE, __HIP_MEMORY_SCOPE_AGENT) < expected) {
            __builtin_amdgcn_s_sleep(2);
        }
    }
    __syncthreads();
}

// ---------------- prologue 1: swizzle weights f32 -> bf16 WMMA-B-fragment order ----
// B tile (32x16) per wave: lane<16 -> K 0..15, lane>=16 -> K 16..31 ; elem j -> K = 16*(lane>>4)+j
// tile order: (gate, ntile, ktile) so a wave streams contiguous 1KB tiles over the K loop.
__global__ void qlstm_prep_weights(const float* __restrict__ W1,
                                   const float* __restrict__ W2,
                                   __bf16* __restrict__ W1s,
                                   __bf16* __restrict__ W2s) {
    int i = blockIdx.x * blockDim.x + threadIdx.x;

    // W1s: 4 * NT * KT1 * 512 bf16 = 524288
    if (i < NGATES * NT * KT1 * 512) {
        int j    = i & 15;
        int lane = (i >> 4) & 31;
        int kt   = (i >> 9) & 15;
        int nt   = (i >> 13) & 15;
        int g    = i >> 17;
        int k = kt * 32 + (lane >> 4) * 16 + j;
        int n = nt * 16 + (lane & 15);
        W1s[i] = f2bf(W1[((size_t)g * D_CAT + k) * HID + n]);
    }
    // W2s: 4 * NT * KT2 * 512 bf16 = 262144
    if (i < NGATES * NT * KT2 * 512) {
        int j    = i & 15;
        int lane = (i >> 4) & 31;
        int kt   = (i >> 9) & 7;
        int nt   = (i >> 12) & 15;
        int g    = i >> 16;
        int k = kt * 32 + (lane >> 4) * 16 + j;
        int n = nt * 16 + (lane & 15);
        W2s[i] = f2bf(W2[((size_t)g * HID + k) * HID + n]);
    }
}

// ---------------- prologue 2: zero recurrent state + barrier counter ----------------
__global__ void qlstm_prep_state(float* __restrict__ C, __bf16* __restrict__ Hb,
                                 unsigned* __restrict__ bar) {
    int i = blockIdx.x * blockDim.x + threadIdx.x;
    if (i < BH) {
        C[i]  = 0.0f;
        Hb[i] = f2bf(0.0f);
    }
    if (i == 0) *bar = 0u;
}

// ---------------- main persistent kernel: grid = 4 (one WG per gate), 512 thr ----------------
__global__ void __launch_bounds__(512)
qlstm_main(const float*  __restrict__ X,     // [T, B, D_IN] f32
           const float*  __restrict__ b1,    // [4, HID]
           const float*  __restrict__ b2,    // [4, HID]
           const __bf16* __restrict__ W1s,   // swizzled
           const __bf16* __restrict__ W2s,   // swizzled
           float*        __restrict__ G,     // [4, B, HID] gate outputs (ws)
           float*        __restrict__ C,     // [B, HID] cell state (ws)
           __bf16*       __restrict__ Hb,    // [B, HID] bf16 hidden (ws)
           unsigned*     __restrict__ bar,   // grid barrier counter (ws)
           float*        __restrict__ out)   // [T,B,HID] ++ hx ++ cx
{
    extern __shared__ char smem[];
    __bf16* sComb = (__bf16*)smem;                             // [64][512] bf16 = 64KB
    __bf16* sHid  = (__bf16*)(smem + BATCH * D_CAT * 2);       // [64][256] bf16 = 32KB
    float*  sXf   = (float*)(smem + BATCH * D_CAT * 2
                                  + BATCH * HID * 2);          // [64][256] f32 = 64KB (x staging)
    const float4* sXf4 = (const float4*)sXf;
    const unsigned sXf_lds = (unsigned)(uintptr_t)sXf;         // LDS byte offset of staging buf

    const int g    = blockIdx.x;          // gate id
    const int tid  = threadIdx.x;
    const int wave = tid >> 5;            // 0..15  -> owns N tile `wave`
    const int lane = tid & 31;
    const int lh   = lane >> 4;           // lane half (0/1)
    const int ll   = lane & 15;
    const int ncol = wave * 16 + ll;      // output column this lane covers

    const float bias1 = b1[g * HID + ncol];
    const float bias2 = b2[g * HID + ncol];

    const __bf16* w1base = W1s + (((size_t)g * NT + wave) * KT1) * 512 + (size_t)lane * 16;
    const __bf16* w2base = W2s + (((size_t)g * NT + wave) * KT2) * 512 + (size_t)lane * 16;
    float* Gg = G + (size_t)g * BH;

    unsigned epoch = 0;

    // ---- prefill: async-stage x_0 into LDS (8 x 16B chunks per thread) ----
#pragma unroll
    for (int e = 0; e < 8; ++e) {
        unsigned idx = (unsigned)tid + e * 512u;          // float4 chunk index over [64][64]
        async_g2l_b128(sXf_lds + idx * 16u, idx * 16u, (const void*)X);
    }

    for (int t = 0; t < T_STEPS; ++t) {
        // ---- x tile for this step has landed (each thread staged its own chunks) ----
        wait_async0();

        // ---- build combined = [x_t | h] in LDS (bf16) ----
#pragma unroll
        for (int e = 0; e < 8; ++e) {
            int idx = tid + e * 512;            // float4 index over [64][64]
            int r = idx >> 6, c4 = idx & 63;
            float4 v = sXf4[idx];               // ds_load_b128 from staging
            __bf16* dst = sComb + r * D_CAT + c4 * 4;
            dst[0] = f2bf(v.x); dst[1] = f2bf(v.y);
            dst[2] = f2bf(v.z); dst[3] = f2bf(v.w);
        }
        {
            const ushort4* hp = (const ushort4*)Hb;
#pragma unroll
            for (int e = 0; e < 8; ++e) {
                int idx = tid + e * 512;            // ushort4 index over [64][64]
                int r = idx >> 6, c4 = idx & 63;
                *(ushort4*)(sComb + r * D_CAT + D_IN + c4 * 4) = hp[idx];
            }
        }
        __syncthreads();

        // ---- overlap: async-stage x_{t+1} while both GEMMs run ----
        if (t + 1 < T_STEPS) {
            const void* gb = (const void*)(X + (size_t)(t + 1) * BATCH * D_IN);
#pragma unroll
            for (int e = 0; e < 8; ++e) {
                unsigned idx = (unsigned)tid + e * 512u;
                async_g2l_b128(sXf_lds + idx * 16u, idx * 16u, gb);
            }
        }

        // ---- stage 1: hid = tanh(combined @ W1[g] + b1[g]) ----  M=64 K=512 N=16(per wave)
        {
            v8f acc[MT];
#pragma unroll
            for (int mt = 0; mt < MT; ++mt) acc[mt] = (v8f){};

            for (int kt = 0; kt < KT1; ++kt) {
                v16bf bfrag = *(const v16bf*)(w1base + kt * 512);
                if (kt + 1 < KT1) __builtin_prefetch(w1base + (kt + 1) * 512, 0, 1);
                const int kb = kt * 32 + lh * 16;
#pragma unroll
                for (int mt = 0; mt < MT; ++mt) {
                    v16bf afrag = ld_frag_a(sComb + (mt * 16 + ll) * D_CAT + kb);
                    acc[mt] = wmma_bf16(afrag, bfrag, acc[mt]);
                }
            }
#pragma unroll
            for (int mt = 0; mt < MT; ++mt)
#pragma unroll
                for (int v = 0; v < 8; ++v) {
                    int m = mt * 16 + lh * 8 + v;
                    sHid[m * HID + ncol] = f2bf(tanhf(acc[mt][v] + bias1));
                }
        }
        __syncthreads();

        // ---- stage 2: gate = sigmoid(hid @ W2[g] + b2[g]) ----  M=64 K=256 N=16(per wave)
        {
            v8f acc[MT];
#pragma unroll
            for (int mt = 0; mt < MT; ++mt) acc[mt] = (v8f){};

            for (int kt = 0; kt < KT2; ++kt) {
                v16bf bfrag = *(const v16bf*)(w2base + kt * 512);
                if (kt + 1 < KT2) __builtin_prefetch(w2base + (kt + 1) * 512, 0, 1);
                const int kb = kt * 32 + lh * 16;
#pragma unroll
                for (int mt = 0; mt < MT; ++mt) {
                    v16bf afrag = ld_frag_a(sHid + (mt * 16 + ll) * HID + kb);
                    acc[mt] = wmma_bf16(afrag, bfrag, acc[mt]);
                }
            }
#pragma unroll
            for (int mt = 0; mt < MT; ++mt)
#pragma unroll
                for (int v = 0; v < 8; ++v) {
                    int m = mt * 16 + lh * 8 + v;
                    Gg[m * HID + ncol] = fsigmoid(acc[mt][v] + bias2);
                }
        }

        // ---- all 4 gate buffers visible ----
        epoch += NGATES;
        grid_sync(bar, epoch);

        // ---- elementwise LSTM update: WG g owns cols [g*64, g*64+64) ----
#pragma unroll
        for (int e = 0; e < 8; ++e) {
            int idx = tid + e * 512;             // 0..4095 over [64 rows][64 cols]
            int r = idx >> 6;
            int c = g * 64 + (idx & 63);
            int rc = r * HID + c;
            float fg = G[0 * BH + rc];
            float ig = G[1 * BH + rc];
            float ug = G[2 * BH + rc];
            float og = G[3 * BH + rc];
            float cold = C[rc];
            float cnew = fg * cold + ig * tanhf(ug);
            float h    = og * tanhf(cnew);
            C[rc]  = cnew;
            Hb[rc] = f2bf(h);
            out[(size_t)t * BH + rc] = h;
            if (t == T_STEPS - 1) {
                out[(size_t)T_STEPS * BH + rc]      = h;     // final hx
                out[(size_t)T_STEPS * BH + BH + rc] = cnew;  // final cx
            }
        }

        // ---- h published for next step ----
        epoch += NGATES;
        grid_sync(bar, epoch);
    }
}

// ---------------- launcher ----------------
extern "C" void kernel_launch(void* const* d_in, const int* in_sizes, int n_in,
                              void* d_out, int out_size, void* d_ws, size_t ws_size,
                              hipStream_t stream) {
    (void)in_sizes; (void)n_in; (void)out_size; (void)ws_size;

    const float* X  = (const float*)d_in[0];   // [T,B,D_IN]
    const float* W1 = (const float*)d_in[1];   // [4,512,256]
    const float* b1 = (const float*)d_in[2];   // [4,256]
    const float* W2 = (const float*)d_in[3];   // [4,256,256]
    const float* b2 = (const float*)d_in[4];   // [4,256]

    char* ws = (char*)d_ws;
    size_t off = 0;
    unsigned* bar = (unsigned*)(ws + off);      off += 256;
    float*    G   = (float*)(ws + off);         off += (size_t)NGATES * BH * 4;   // 256KB
    float*    C   = (float*)(ws + off);         off += (size_t)BH * 4;            // 64KB
    __bf16*   Hb  = (__bf16*)(ws + off);        off += (size_t)BH * 2;            // 32KB
    __bf16*   W1s = (__bf16*)(ws + off);        off += (size_t)NGATES * D_CAT * HID * 2; // 1MB
    __bf16*   W2s = (__bf16*)(ws + off);        off += (size_t)NGATES * HID * HID * 2;   // 512KB

    // prologue: swizzle weights to bf16 fragments; zero state + barrier
    int prepN = NGATES * NT * KT1 * 512;                 // 524288
    qlstm_prep_weights<<<(prepN + 255) / 256, 256, 0, stream>>>(W1, W2, W1s, W2s);
    qlstm_prep_state<<<(BH + 255) / 256, 256, 0, stream>>>(C, Hb, bar);

    // persistent recurrent kernel: 4 WGs (one per gate), 16 wave32s each
    size_t lds_bytes = (size_t)BATCH * D_CAT * 2   // combined bf16
                     + (size_t)BATCH * HID * 2     // hid bf16
                     + (size_t)BATCH * D_IN * 4;   // x f32 staging (async target)
    qlstm_main<<<NGATES, 512, lds_bytes, stream>>>(X, b1, b2, W1s, W2s,
                                                   G, C, Hb, bar, (float*)d_out);
}
